// GLSTMseq2seq_42382737277019
// MI455X (gfx1250) — matrix operand order repaired and checked by reading, not compiled, not run on hardware
//
#include <hip/hip_runtime.h>
#include <hip/hip_bf16.h>

// ---------------------------------------------------------------------------
// GLSTM seq2seq for gfx1250 (MI455X): bf16 WMMA (v_wmma_f32_16x16x32_bf16)
// for every matmul, f32 accumulation, fused attention in 224KB LDS.
// GEMM uses double-buffered LDS staging: next tile's global loads are issued
// before the current tile's WMMAs, stores drain after; one barrier per k-step.
// ---------------------------------------------------------------------------

typedef __bf16 bf16_t;
typedef __attribute__((ext_vector_type(16))) __bf16 v16bf;
typedef __attribute__((ext_vector_type(8)))  float  v8f;

#define B_    16
#define P_    24
#define FUT_  12
#define N_    256
#define CONT_ 5
#define DH_   128
#define KPRE_ 224   // 197 padded to multiple of 32
#define KFUT_ 352   // 321 padded to multiple of 32
#define ROWS_ALL (B_ * P_ * N_)   // 98304
#define ROWS_BN  (B_ * N_)        // 4096

union FragU { uint4 u[2]; v16bf v; };

// A fragment 16x32 bf16 from row-major LDS (ld elems per row, 16B aligned
// rows). lane m<16 -> row m; elements [0..15] = contiguous K 16*hi..16*hi+15
__device__ __forceinline__ v16bf frag_a_lds(const bf16_t* base, int ld, int lane) {
  const int m = lane & 15, hi = lane >> 4;
  const bf16_t* row = base + (size_t)m * ld + 16 * hi;
  FragU t;
  t.u[0] = *(const uint4*)(row);
  t.u[1] = *(const uint4*)(row + 8);
  return t.v;
}

// B fragment 32x16 bf16 from TRANSPOSED LDS staging baseT[n][k] (ldT elems
// per n-row): lane n reads contiguous K = 16*hi .. 16*hi+15.
__device__ __forceinline__ v16bf frag_b_swzT(const bf16_t* baseT, int ldT, int lane) {
  const int n = lane & 15, hi = lane >> 4;
  const bf16_t* p = baseT + (size_t)n * ldT + 16 * hi;
  FragU t;
  t.u[0] = *(const uint4*)(p);
  t.u[1] = *(const uint4*)(p + 8);
  return t.v;
}

__device__ __forceinline__ float sigf(float x) {
  return __builtin_amdgcn_rcpf(1.f + __expf(-x));
}

// ---------------- Generic bf16 WMMA GEMM -----------------------------------
// C[M,N] = act(A[M,K] @ W[K,N] + bias) ; A row-major bf16, WT = W transposed
// (N x K) bf16, f32 accum. Block tile 128x128, 8 waves; wave = 2 row-tiles x
// 4 col-tiles (8 wmma/k-step). M,N mult of 128, K mult of 32.
// Double-buffered LDS staging. act: 0=none 1=relu 2=sigmoid.
__global__ __launch_bounds__(256) void gemm_bf16_wmma(
    const bf16_t* __restrict__ A, const bf16_t* __restrict__ WT,
    const float* __restrict__ bias, float* __restrict__ Cf,
    bf16_t* __restrict__ Cb, int M, int N, int K, int act)
{
  __shared__ bf16_t As [2][128][32];   // 16KB ping-pong, rows of A
  __shared__ bf16_t WsT[2][128][32];   // 16KB ping-pong, WsT[n][k]
  const int tid = threadIdx.x, lane = tid & 31, wv = tid >> 5;
  const int wm = wv & 3, wn = wv >> 2;     // wave rows [wm*32,+32), cols [wn*64,+64)
  const int bm = blockIdx.x * 128, bn = blockIdx.y * 128;

  const v8f vzero = {0.f, 0.f, 0.f, 0.f, 0.f, 0.f, 0.f, 0.f};
  v8f acc[2][4] = {{vzero, vzero, vzero, vzero}, {vzero, vzero, vzero, vzero}};

  const int r  = tid >> 1;            // 0..127
  const int c0 = (tid & 1) * 16;      // 0 or 16
  const bf16_t* srcA = A  + (size_t)(bm + r) * K + c0;
  const bf16_t* srcW = WT + (size_t)(bn + r) * K + c0;

  // prologue: stage first tile into buffer 0
  {
    uint4 ra0 = *(const uint4*)(srcA);
    uint4 ra1 = *(const uint4*)(srcA + 8);
    uint4 rw0 = *(const uint4*)(srcW);
    uint4 rw1 = *(const uint4*)(srcW + 8);
    *(uint4*)&As [0][r][c0]     = ra0;
    *(uint4*)&As [0][r][c0 + 8] = ra1;
    *(uint4*)&WsT[0][r][c0]     = rw0;
    *(uint4*)&WsT[0][r][c0 + 8] = rw1;
  }
  __syncthreads();

  for (int k0 = 0; k0 < K; k0 += 32) {
    const int cur = (k0 >> 5) & 1, nxt = cur ^ 1;
    const bool more = (k0 + 32 < K);
    // issue next tile's global loads BEFORE the matrix math
    uint4 ra0, ra1, rw0, rw1;
    if (more) {
      ra0 = *(const uint4*)(srcA + k0 + 32);
      ra1 = *(const uint4*)(srcA + k0 + 40);
      rw0 = *(const uint4*)(srcW + k0 + 32);
      rw1 = *(const uint4*)(srcW + k0 + 40);
      if (k0 + 64 < K) {
        __builtin_prefetch(srcA + k0 + 64, 0, 3);   // global_prefetch_b8
        __builtin_prefetch(srcW + k0 + 64, 0, 3);
      }
    }
    // compute on current buffer
    const v16bf a0 = frag_a_lds(&As[cur][wm * 32][0],      32, lane);
    const v16bf a1 = frag_a_lds(&As[cur][wm * 32 + 16][0], 32, lane);
#pragma unroll
    for (int j = 0; j < 4; ++j) {
      const v16bf b = frag_b_swzT(&WsT[cur][wn * 64 + j * 16][0], 32, lane);
      acc[0][j] = __builtin_amdgcn_wmma_f32_16x16x32_bf16(false, a0, false, b, (short)0, acc[0][j], false, false);
      acc[1][j] = __builtin_amdgcn_wmma_f32_16x16x32_bf16(false, a1, false, b, (short)0, acc[1][j], false, false);
    }
    // drain next tile into the other buffer
    if (more) {
      *(uint4*)&As [nxt][r][c0]     = ra0;
      *(uint4*)&As [nxt][r][c0 + 8] = ra1;
      *(uint4*)&WsT[nxt][r][c0]     = rw0;
      *(uint4*)&WsT[nxt][r][c0 + 8] = rw1;
    }
    __syncthreads();
  }
  // epilogue: C layout — VGPR r, lane l: row = r + 8*(l>=16), col = l&15
  const int nl = lane & 15, hi = lane >> 4;
#pragma unroll
  for (int j = 0; j < 4; ++j) {
    const int gn = bn + wn * 64 + j * 16 + nl;
    const float bv = bias ? bias[gn] : 0.f;
#pragma unroll
    for (int i = 0; i < 2; ++i)
#pragma unroll
      for (int rr = 0; rr < 8; ++rr) {
        const int gm = bm + wm * 32 + i * 16 + rr + 8 * hi;
        float v = acc[i][j][rr] + bv;
        if (act == 1) v = fmaxf(v, 0.f);
        else if (act == 2) v = sigf(v);
        const size_t off = (size_t)gm * N + gn;
        if (Cf) Cf[off] = v;
        if (Cb) Cb[off] = (bf16_t)v;
      }
  }
}

// ---------------- Fused GNN attention --------------------------------------
// One workgroup per (b*P+p, 64-row strip). Computes
//   S = xe_strip @ xe^T ; mask(adj) ; softmax ; out = P @ x   (all in LDS)
// Dynamic LDS: xe 64KB | S 64KB f32 | P 32KB bf16 | x^T 64KB  = 224KB
__global__ __launch_bounds__(256) void attn_kernel(
    const bf16_t* __restrict__ xe, const bf16_t* __restrict__ xv,
    const float* __restrict__ adj, bf16_t* __restrict__ g_out)
{
  extern __shared__ char smem[];
  bf16_t* xeS  = (bf16_t*)smem;                 // 256*128 bf16, row-major
  float*  S    = (float*)(smem + 65536);        // 64*256 f32
  bf16_t* Pb   = (bf16_t*)(smem + 131072);      // 64*256 bf16
  bf16_t* xS_T = (bf16_t*)(smem + 163840);      // x TRANSPOSED: [feat 128][node 256]
  const int tid = threadIdx.x, lane = tid & 31, wv = tid >> 5;
  const int bp = blockIdx.x, r0 = blockIdx.y * 64;
  const bf16_t* xe_g = xe + (size_t)bp * N_ * DH_;
  const bf16_t* xv_g = xv + (size_t)bp * N_ * DH_;

  for (int i = tid; i < N_ * DH_ / 8; i += 256)
    ((uint4*)xeS)[i] = ((const uint4*)xe_g)[i];
  __syncthreads();

  const int rt = wv & 3;          // 16-row tile within strip
  const int nl = lane & 15, hi = lane >> 4;

  // ---- pass 1: S_strip(64x256) = xe[r0:r0+64] @ xe^T ----
  {
    const int cb = (wv >> 2) * 8;   // 8 col tiles per wave
    const v8f vzero = {0.f, 0.f, 0.f, 0.f, 0.f, 0.f, 0.f, 0.f};
    v8f acc[8] = {vzero, vzero, vzero, vzero, vzero, vzero, vzero, vzero};
    for (int k0 = 0; k0 < DH_; k0 += 32) {
      const v16bf af = frag_a_lds(xeS + (size_t)(r0 + rt * 16) * DH_ + k0, DH_, lane);
#pragma unroll
      for (int t = 0; t < 8; ++t) {
        // B = xe^T: col n of B is row n of xe -> contiguous 32B per lane
        const bf16_t* rp = xeS + (size_t)((cb + t) * 16 + nl) * DH_ + k0 + 16 * hi;
        FragU tb;
        tb.u[0] = *(const uint4*)(rp);
        tb.u[1] = *(const uint4*)(rp + 8);
        acc[t] = __builtin_amdgcn_wmma_f32_16x16x32_bf16(false, af, false, tb.v, (short)0, acc[t], false, false);
      }
    }
#pragma unroll
    for (int t = 0; t < 8; ++t)
#pragma unroll
      for (int rr = 0; rr < 8; ++rr) {
        const int li = rt * 16 + rr + 8 * hi;
        const int gj = (cb + t) * 16 + nl;
        const float a = adj[(size_t)(r0 + li) * N_ + gj];
        S[li * N_ + gj] = (a == 0.f) ? -__builtin_inff() : acc[t][rr];
      }
  }
  __syncthreads();

  // ---- softmax rows (threads 0..63); others stage x transposed into LDS ----
  if (tid < 64) {
    float* row = S + tid * N_;
    float m = -__builtin_inff();
    for (int j = 0; j < N_; ++j) m = fmaxf(m, row[j]);
    float s = 0.f;
    for (int j = 0; j < N_; ++j) { const float e = __expf(row[j] - m); row[j] = e; s += e; }
    const float inv = 1.f / s;
    bf16_t* prow = Pb + tid * N_;
    for (int j = 0; j < N_; ++j) prow[j] = (bf16_t)(row[j] * inv);
  } else {
    for (int i = tid - 64; i < N_ * DH_ / 8; i += 192) {
      const int node = i >> 4;             // 16 x uint4 chunks per node row
      const int c0 = (i & 15) * 8;         // feature start
      uint4 wv4 = ((const uint4*)xv_g)[i];
      const bf16_t* wp = (const bf16_t*)&wv4;
#pragma unroll
      for (int jj = 0; jj < 8; ++jj) xS_T[(size_t)(c0 + jj) * N_ + node] = wp[jj];
    }
  }
  __syncthreads();

  // ---- pass 2: out(64x128) = P(64x256) @ x(256x128) ----
  {
    const int cb = (wv >> 2) * 4;   // 4 col (feature) tiles per wave
    const v8f vzero = {0.f, 0.f, 0.f, 0.f, 0.f, 0.f, 0.f, 0.f};
    v8f acc[4] = {vzero, vzero, vzero, vzero};
    for (int k0 = 0; k0 < N_; k0 += 32) {
      const v16bf af = frag_a_lds(Pb + (size_t)(rt * 16) * N_ + k0, N_, lane);
#pragma unroll
      for (int t = 0; t < 4; ++t) {
        const v16bf b = frag_b_swzT(xS_T + (size_t)((cb + t) * 16) * N_ + k0, N_, lane);
        acc[t] = __builtin_amdgcn_wmma_f32_16x16x32_bf16(false, af, false, b, (short)0, acc[t], false, false);
      }
    }
#pragma unroll
    for (int t = 0; t < 4; ++t)
#pragma unroll
      for (int rr = 0; rr < 8; ++rr) {
        const int li = rt * 16 + rr + 8 * hi;
        const int gn = (cb + t) * 16 + nl;
        g_out[((size_t)bp * N_ + r0 + li) * DH_ + gn] = (bf16_t)acc[t][rr];
      }
  }
}

// ---------------- Elementwise / packing kernels ----------------------------
__global__ void embed_kernel(const float* __restrict__ x_cont, const int* __restrict__ x_cat,
                             const float* __restrict__ e0, const float* __restrict__ e1,
                             const float* __restrict__ e2, bf16_t* __restrict__ emb)
{
  const long idx = (long)blockIdx.x * blockDim.x + threadIdx.x;
  if (idx >= (long)ROWS_ALL * KPRE_) return;
  const int row = (int)(idx / KPRE_), d = (int)(idx % KPRE_);
  float v = 0.f;
  if (d < CONT_) v = x_cont[(size_t)row * CONT_ + d];
  else if (d < 197) {
    const int dd = d - CONT_, l = dd >> 6, off = dd & 63;
    const float* tab = (l == 0) ? e0 : (l == 1) ? e1 : e2;
    v = tab[(size_t)x_cat[(size_t)row * 3 + l] * 64 + off];
  }
  emb[(size_t)row * KPRE_ + d] = (bf16_t)v;
}

__global__ void gather_embfut_kernel(const bf16_t* __restrict__ emb, bf16_t* __restrict__ ef)
{
  const int idx = blockIdx.x * blockDim.x + threadIdx.x;
  if (idx >= ROWS_BN * 192) return;
  const int row = idx / 192, d = idx % 192;
  const int b = row >> 8, n = row & 255;   // p = 0
  ef[idx] = emb[((size_t)b * P_ * N_ + n) * KPRE_ + CONT_ + d];
}

// TRANSPOSING pack: dst is W^T (Ndst x Kdst): dst[col0+n][k] = src[k][n] or 0.
__global__ void pack_w_kernel(const float* __restrict__ src, bf16_t* __restrict__ dst,
                              int Ksrc, int Ncols, int Kdst, int col0)
{
  const int idx = blockIdx.x * blockDim.x + threadIdx.x;
  if (idx >= Kdst * Ncols) return;
  const int k = idx / Ncols, n = idx % Ncols;
  const float v = (k < Ksrc) ? src[(size_t)k * Ncols + n] : 0.f;
  dst[(size_t)(col0 + n) * Kdst + k] = (bf16_t)v;
}

__global__ void pack_bias4_kernel(const float* __restrict__ b0, const float* __restrict__ b1,
                                  const float* __restrict__ b2, const float* __restrict__ b3,
                                  float* __restrict__ dst)
{
  const int i = blockIdx.x * blockDim.x + threadIdx.x;
  if (i >= 512) return;
  const float* s = (i < 128) ? b0 : (i < 256) ? b1 : (i < 384) ? b2 : b3;
  dst[i] = s[i & 127];
}

__global__ void concat_past_kernel(const bf16_t* __restrict__ xg, const bf16_t* __restrict__ hb,
                                   bf16_t* __restrict__ Acat, int t)
{
  const int idx = blockIdx.x * blockDim.x + threadIdx.x;
  if (idx >= ROWS_BN * 256) return;
  const int row = idx >> 8, d = idx & 255;
  const int b = row >> 8, n = row & 255;
  bf16_t v;
  if (d < 128) v = xg[((size_t)(b * P_ + t) * N_ + n) * DH_ + d];
  else         v = hb[(size_t)row * DH_ + (d - 128)];
  Acat[(size_t)row * 256 + d] = v;
}

__global__ void concat_fut_kernel(const float* __restrict__ prev, const bf16_t* __restrict__ ef,
                                  const bf16_t* __restrict__ hb, bf16_t* __restrict__ Afut)
{
  const long idx = (long)blockIdx.x * blockDim.x + threadIdx.x;
  if (idx >= (long)ROWS_BN * KFUT_) return;
  const int row = (int)(idx / KFUT_), d = (int)(idx % KFUT_);
  bf16_t v;
  if (d == 0)        v = (bf16_t)prev[row];
  else if (d < 193)  v = ef[(size_t)row * 192 + (d - 1)];
  else if (d < 321)  v = hb[(size_t)row * DH_ + (d - 193)];
  else               v = (bf16_t)0.f;
  Afut[(size_t)row * KFUT_ + d] = v;
}

// gates layout: [i | f | o | g] each 128 wide; h/c state in f32
__global__ void lstm_cell_kernel(const float* __restrict__ gates, float* __restrict__ c,
                                 float* __restrict__ h, bf16_t* __restrict__ hb)
{
  const int idx = blockIdx.x * blockDim.x + threadIdx.x;
  if (idx >= ROWS_BN * DH_) return;
  const int row = idx >> 7, j = idx & 127;
  const float* gr = gates + (size_t)row * 512;
  const float ig = sigf(gr[j]);
  const float fg = sigf(gr[128 + j]);
  const float og = sigf(gr[256 + j]);
  const float gg = tanhf(gr[384 + j]);
  const float cn = fg * c[idx] + ig * gg;
  c[idx] = cn;
  const float hn = og * tanhf(cn);
  h[idx]  = hn;
  hb[idx] = (bf16_t)hn;
}

// final decoder layer 128 -> 1 ; writes d_out[(b*FUT+t)*N + n] and prev[row]
__global__ void dec3_kernel(const bf16_t* __restrict__ d2v, const float* __restrict__ W3,
                            const float* __restrict__ b3, float* __restrict__ out,
                            float* __restrict__ prev, int t)
{
  const int row = blockIdx.x * blockDim.x + threadIdx.x;
  if (row >= ROWS_BN) return;
  float s = b3[0];
  const bf16_t* r = d2v + (size_t)row * DH_;
#pragma unroll 8
  for (int k = 0; k < DH_; ++k) s += (float)r[k] * W3[k];
  const int b = row >> 8, n = row & 255;
  out[((size_t)b * FUT_ + t) * N_ + n] = s;
  prev[row] = s;
}

// ---------------------------------------------------------------------------
extern "C" void kernel_launch(void* const* d_in, const int* in_sizes, int n_in,
                              void* d_out, int out_size, void* d_ws, size_t ws_size,
                              hipStream_t stream)
{
  (void)in_sizes; (void)n_in; (void)out_size; (void)ws_size;
  // pytree leaf order: x_cont, x_cat, adj, then params with sorted dict keys:
  // dec, emb, gnn_emb, gnn_lin, lstm, lstm_fut, pre
  const float* x_cont = (const float*)d_in[0];
  const int*   x_cat  = (const int*)d_in[1];
  const float* adj    = (const float*)d_in[2];
  const float* decW1  = (const float*)d_in[3];
  const float* decb1  = (const float*)d_in[4];
  const float* decW2  = (const float*)d_in[5];
  const float* decb2  = (const float*)d_in[6];
  const float* decW3  = (const float*)d_in[7];
  const float* decb3  = (const float*)d_in[8];
  const float* emb0   = (const float*)d_in[9];
  const float* emb1   = (const float*)d_in[10];
  const float* emb2   = (const float*)d_in[11];
  const float* gnnEmb = (const float*)d_in[12];
  const float* gnnLin = (const float*)d_in[13];
  const float* lstmW[4] = {(const float*)d_in[14], (const float*)d_in[16],
                           (const float*)d_in[18], (const float*)d_in[20]};
  const float* lstmB[4] = {(const float*)d_in[15], (const float*)d_in[17],
                           (const float*)d_in[19], (const float*)d_in[21]};
  const float* futW[4]  = {(const float*)d_in[22], (const float*)d_in[24],
                           (const float*)d_in[26], (const float*)d_in[28]};
  const float* futB[4]  = {(const float*)d_in[23], (const float*)d_in[25],
                           (const float*)d_in[27], (const float*)d_in[29]};
  const float* preW1 = (const float*)d_in[30];
  const float* preB1 = (const float*)d_in[31];
  const float* preW2 = (const float*)d_in[32];
  const float* preB2 = (const float*)d_in[33];
  const float* preW3 = (const float*)d_in[34];
  float* out = (float*)d_out;

  // bump allocator over workspace
  char* p = (char*)d_ws;
  auto alloc = [&](size_t bytes) -> char* {
    char* r = p; p += (bytes + 255) & ~(size_t)255; return r;
  };
  bf16_t* emb    = (bf16_t*)alloc((size_t)ROWS_ALL * KPRE_ * 2);
  bf16_t* h1     = (bf16_t*)alloc((size_t)ROWS_ALL * DH_ * 2);
  bf16_t* h2     = (bf16_t*)alloc((size_t)ROWS_ALL * DH_ * 2);
  bf16_t* x3     = (bf16_t*)alloc((size_t)ROWS_ALL * DH_ * 2);
  bf16_t* xe     = (bf16_t*)alloc((size_t)ROWS_ALL * DH_ * 2);
  bf16_t* g      = h1;   // reuse: h1 dead after x3
  bf16_t* xg     = h2;   // reuse: h2 dead after x3
  bf16_t* embfut = (bf16_t*)alloc((size_t)ROWS_BN * 192 * 2);
  // all weight buffers hold W^T (N x K)
  bf16_t* W1p  = (bf16_t*)alloc((size_t)DH_ * KPRE_ * 2);
  bf16_t* W2b  = (bf16_t*)alloc((size_t)DH_ * DH_ * 2);
  bf16_t* W3b  = (bf16_t*)alloc((size_t)DH_ * DH_ * 2);
  bf16_t* geb  = (bf16_t*)alloc((size_t)DH_ * DH_ * 2);
  bf16_t* glb  = (bf16_t*)alloc((size_t)DH_ * DH_ * 2);
  bf16_t* dW1b = (bf16_t*)alloc((size_t)DH_ * DH_ * 2);
  bf16_t* dW2b = (bf16_t*)alloc((size_t)DH_ * DH_ * 2);
  bf16_t* WcatP = (bf16_t*)alloc((size_t)512 * 256 * 2);
  bf16_t* WcatF = (bf16_t*)alloc((size_t)512 * KFUT_ * 2);
  float*  biasP = (float*)alloc(512 * 4);
  float*  biasF = (float*)alloc(512 * 4);
  bf16_t* Acat  = (bf16_t*)alloc((size_t)ROWS_BN * 256 * 2);
  bf16_t* Afut  = (bf16_t*)alloc((size_t)ROWS_BN * KFUT_ * 2);
  float*  gates = (float*)alloc((size_t)ROWS_BN * 512 * 4);
  float*  hS    = (float*)alloc((size_t)ROWS_BN * DH_ * 4);
  float*  cS    = (float*)alloc((size_t)ROWS_BN * DH_ * 4);
  bf16_t* hb    = (bf16_t*)alloc((size_t)ROWS_BN * DH_ * 2);
  bf16_t* d1    = (bf16_t*)alloc((size_t)ROWS_BN * DH_ * 2);
  bf16_t* d2    = (bf16_t*)alloc((size_t)ROWS_BN * DH_ * 2);
  float*  prev  = (float*)alloc((size_t)ROWS_BN * 4);

  auto packw = [&](const float* s, bf16_t* dst, int Ksrc, int Ncols, int Kdst, int col0) {
    const int total = Kdst * Ncols;
    pack_w_kernel<<<(total + 255) / 256, 256, 0, stream>>>(s, dst, Ksrc, Ncols, Kdst, col0);
  };
  auto gemm = [&](const bf16_t* A, const bf16_t* WT, const float* bias,
                  float* Cf, bf16_t* Cb, int M, int N, int K, int act) {
    gemm_bf16_wmma<<<dim3(M / 128, N / 128), 256, 0, stream>>>(A, WT, bias, Cf, Cb, M, N, K, act);
  };

  // ---- weight repack to transposed bf16 (K padded) ----
  packw(preW1, W1p, 197, 128, KPRE_, 0);
  packw(preW2, W2b, 128, 128, 128, 0);
  packw(preW3, W3b, 128, 128, 128, 0);
  packw(gnnEmb, geb, 128, 128, 128, 0);
  packw(gnnLin, glb, 128, 128, 128, 0);
  packw(decW1, dW1b, 128, 128, 128, 0);
  packw(decW2, dW2b, 128, 128, 128, 0);
  for (int gi = 0; gi < 4; ++gi) {
    packw(lstmW[gi], WcatP, 256, 128, 256, gi * 128);
    packw(futW[gi],  WcatF, 321, 128, KFUT_, gi * 128);
  }
  pack_bias4_kernel<<<2, 256, 0, stream>>>(lstmB[0], lstmB[1], lstmB[2], lstmB[3], biasP);
  pack_bias4_kernel<<<2, 256, 0, stream>>>(futB[0], futB[1], futB[2], futB[3], biasF);

  // ---- embeddings + pre-MLP ----
  {
    const long total = (long)ROWS_ALL * KPRE_;
    embed_kernel<<<(unsigned)((total + 255) / 256), 256, 0, stream>>>(x_cont, x_cat, emb0, emb1, emb2, emb);
  }
  gather_embfut_kernel<<<(ROWS_BN * 192 + 255) / 256, 256, 0, stream>>>(emb, embfut);

  gemm(emb, W1p, preB1, nullptr, h1, ROWS_ALL, 128, KPRE_, 1);  // relu
  gemm(h1,  W2b, preB2, nullptr, h2, ROWS_ALL, 128, 128, 1);    // relu
  gemm(h2,  W3b, nullptr, nullptr, x3, ROWS_ALL, 128, 128, 0);
  gemm(x3,  geb, nullptr, nullptr, xe, ROWS_ALL, 128, 128, 0);

  // ---- fused attention (softmax(xe xe^T masked by adj) @ x3) ----
  attn_kernel<<<dim3(B_ * P_, N_ / 64), 256, 229376, stream>>>(xe, x3, adj, g);
  gemm(g, glb, nullptr, nullptr, xg, ROWS_ALL, 128, 128, 2);    // sigmoid

  // ---- past LSTM over P_ steps ----
  hipMemsetAsync(hS, 0, (size_t)ROWS_BN * DH_ * 4, stream);
  hipMemsetAsync(cS, 0, (size_t)ROWS_BN * DH_ * 4, stream);
  hipMemsetAsync(hb, 0, (size_t)ROWS_BN * DH_ * 2, stream);
  for (int t = 0; t < P_; ++t) {
    concat_past_kernel<<<(ROWS_BN * 256 + 255) / 256, 256, 0, stream>>>(xg, hb, Acat, t);
    gemm(Acat, WcatP, biasP, gates, nullptr, ROWS_BN, 512, 256, 0);
    lstm_cell_kernel<<<(ROWS_BN * DH_ + 255) / 256, 256, 0, stream>>>(gates, cS, hS, hb);
  }

  // ---- decode out0 (t = 0) ----
  gemm(hb, dW1b, decb1, nullptr, d1, ROWS_BN, 128, 128, 1);
  gemm(d1, dW2b, decb2, nullptr, d2, ROWS_BN, 128, 128, 1);
  dec3_kernel<<<(ROWS_BN + 255) / 256, 256, 0, stream>>>(d2, decW3, decb3, out, prev, 0);

  // ---- future LSTM, FUT_-1 steps ----
  hipMemsetAsync(hS, 0, (size_t)ROWS_BN * DH_ * 4, stream);
  hipMemsetAsync(cS, 0, (size_t)ROWS_BN * DH_ * 4, stream);
  hipMemsetAsync(hb, 0, (size_t)ROWS_BN * DH_ * 2, stream);
  for (int s = 1; s < FUT_; ++s) {
    const long totalF = (long)ROWS_BN * KFUT_;
    concat_fut_kernel<<<(unsigned)((totalF + 255) / 256), 256, 0, stream>>>(prev, embfut, hb, Afut);
    gemm(Afut, WcatF, biasF, gates, nullptr, ROWS_BN, 512, KFUT_, 0);
    lstm_cell_kernel<<<(ROWS_BN * DH_ + 255) / 256, 256, 0, stream>>>(gates, cS, hS, hb);
    gemm(hb, dW1b, decb1, nullptr, d1, ROWS_BN, 128, 128, 1);
    gemm(d1, dW2b, decb2, nullptr, d2, ROWS_BN, 128, 128, 1);
    dec3_kernel<<<(ROWS_BN + 255) / 256, 256, 0, stream>>>(d2, decW3, decb3, out, prev, s);
  }
}
